// ForwardLSTM_47467978555644
// MI455X (gfx1250) — matrix operand order repaired and checked by reading, not compiled, not run on hardware
//
#include <hip/hip_runtime.h>
#include <hip/hip_bf16.h>

// ---------------------------------------------------------------------------
// ForwardLSTM for MI455X (gfx1250):
//   K1: xWi = x @ Wi         -> WMMA bf16-split (3x v_wmma_f32_16x16x32_bf16 / K-tile)
//   K2: sequential LSTM scan -> single WGP; Wh as packed bf16: 52 VGPRs + 304KB LDS,
//                               v_fma_mix_f32_bf16 (1 instr / MAC, f32 accumulate,
//                               4 independent accumulator chains)
//   K3: LayerNorm + ReLU + (256->3) head -> wave-per-row, shfl_xor reductions
// ---------------------------------------------------------------------------

#define T_LEN 32768
#define F_DIM 256
#define G_DIM 1024   // 4*F

#define RREG_ROWS 104                 // Wh rows 0..103 in VGPRs (52 packed dwords)
#define RLDS_ROWS 152                 // Wh rows 104..255 in LDS (76 packed dwords/col)
#define WLDS_DWORDS ((RLDS_ROWS / 2) * G_DIM)          // 77824 dwords = 311296 B
#define SMEM_BYTES (WLDS_DWORDS * 4 + F_DIM * 4 + G_DIM * 4)  // + hsh + zsh = 316416 B

typedef __attribute__((ext_vector_type(16))) __bf16 bf16x16;
typedef __attribute__((ext_vector_type(8)))  float  f32x8;

static __device__ inline unsigned f2u(float f) { union { float f; unsigned u; } v; v.f = f; return v.u; }
static __device__ inline float    u2f(unsigned u) { union { unsigned u; float f; } v; v.u = u; return v.f; }
static __device__ inline __bf16   bfbits(unsigned short s) { union { unsigned short s; __bf16 b; } v; v.s = s; return v.b; }

// f32 -> bf16 bits, round-to-nearest-even
static __device__ inline unsigned bf16_rne(float x) {
    unsigned u = f2u(x);
    unsigned r = (u + 0x7FFFu + ((u >> 16) & 1u)) >> 16;
    return r & 0xFFFFu;
}
// pack two rows: lo16 = row k (even), hi16 = row k+1 (odd)
static __device__ inline unsigned pack_bf16(float lo, float hi) {
    return bf16_rne(lo) | (bf16_rne(hi) << 16);
}

// Split an f32 into bf16 hi + bf16 lo terms (truncation split, ~16 mantissa bits)
static __device__ inline void split_bf(float x, __bf16& hi, __bf16& lo) {
    unsigned u = f2u(x);
    unsigned short hb = (unsigned short)(u >> 16);
    float hf = u2f(((unsigned)hb) << 16);
    float l  = x - hf;
    unsigned short lb = (unsigned short)(f2u(l) >> 16);
    hi = bfbits(hb);
    lo = bfbits(lb);
}

// acc += h * bf16(lo 16 bits of p)   (VOP3P v_fma_mix_f32_bf16, f32 accumulate)
static __device__ inline void mix_lo(float& acc, float h, unsigned p) {
    asm volatile("v_fma_mix_f32_bf16 %0, %1, %2, %0 op_sel:[0,0,0] op_sel_hi:[0,1,0]"
                 : "+v"(acc) : "v"(h), "v"(p));
}
// acc += h * bf16(hi 16 bits of p)
static __device__ inline void mix_hi(float& acc, float h, unsigned p) {
    asm volatile("v_fma_mix_f32_bf16 %0, %1, %2, %0 op_sel:[0,1,0] op_sel_hi:[0,1,0]"
                 : "+v"(acc) : "v"(h), "v"(p));
}

// ---------------------------------------------------------------------------
// Kernel 1: xWi = x @ Wi   (32768x256 @ 256x1024, f32 in/out)
// One wave per 16x16 output tile; bf16 split => 3 WMMAs / 32-wide K tile.
// ---------------------------------------------------------------------------
__global__ void gemm_xwi_wmma(const float* __restrict__ X,
                              const float* __restrict__ Wi,
                              float* __restrict__ XWI) {
    const int wavesPerBlock = blockDim.x >> 5;
    const int wave = blockIdx.x * wavesPerBlock + (threadIdx.x >> 5);
    const int lane = threadIdx.x & 31;
    const int NT = G_DIM / 16;          // 64 tiles along N
    const int mTile = wave / NT;
    const int nTile = wave % NT;
    if (mTile >= T_LEN / 16) return;

    const int half = lane >> 4;         // 0: lanes 0-15, 1: lanes 16-31
    const int mn   = lane & 15;         // M row (A) / N col (B,C)
    const int m0 = mTile * 16;
    const int n0 = nTile * 16;

    f32x8 acc = {0.f, 0.f, 0.f, 0.f, 0.f, 0.f, 0.f, 0.f};

    #pragma unroll
    for (int kb = 0; kb < F_DIM / 32; ++kb) {
        bf16x16 ah, al, bhv, blv;

        // A fragment: 16x32 bf16; lane holds row m0+mn.
        // element e -> K = (e<8 ? 0 : 16) + half*8 + (e&7)   (ISA 7.12.2)
        const float* xr = X + (size_t)(m0 + mn) * F_DIM + kb * 32;
        #pragma unroll
        for (int e = 0; e < 16; ++e) {
            int k = ((e < 8) ? 0 : 16) + half * 8 + (e & 7);
            __bf16 h, l; split_bf(xr[k], h, l);
            ah[e] = h; al[e] = l;
        }
        // B fragment: 32x16 bf16; lane holds col n0+mn; element e -> K = (e<8 ? 0 : 16) + half*8 + (e&7)
        #pragma unroll
        for (int e = 0; e < 16; ++e) {
            int k = kb * 32 + ((e < 8) ? 0 : 16) + half * 8 + (e & 7);
            __bf16 h, l; split_bf(Wi[(size_t)k * G_DIM + n0 + mn], h, l);
            bhv[e] = h; blv[e] = l;
        }

        acc = __builtin_amdgcn_wmma_f32_16x16x32_bf16(false, al, false, bhv, (short)0, acc, false, false);
        acc = __builtin_amdgcn_wmma_f32_16x16x32_bf16(false, ah, false, blv, (short)0, acc, false, false);
        acc = __builtin_amdgcn_wmma_f32_16x16x32_bf16(false, ah, false, bhv, (short)0, acc, false, false);
    }

    // C/D layout: VGPR r -> M = r + half*8, N = mn
    float* out = XWI + (size_t)(m0 + half * 8) * G_DIM + n0 + mn;
    #pragma unroll
    for (int r = 0; r < 8; ++r) out[(size_t)r * G_DIM] = acc[r];
}

// ---------------------------------------------------------------------------
// Kernel 2: sequential LSTM scan. Single 1024-thread workgroup on one WGP.
// Thread j owns gate column j. Wh column j (256 entries) kept as packed bf16:
//   rows 0..103   -> 52 VGPRs (no spills at full-WGP occupancy)
//   rows 104..255 -> LDS, row-major (lane j -> bank j, conflict free), 304 KB
// MACs via v_fma_mix_f32_bf16 into 4 independent f32 accumulators (breaks the
// serial FMA chain). xWi row software-pipelined one step ahead +
// global_prefetch_b8 two ahead.
// ---------------------------------------------------------------------------
__global__ __launch_bounds__(1024) void lstm_scan(
        const float* __restrict__ XWI, const float* __restrict__ Wh,
        const float* __restrict__ bh,  const float* __restrict__ c0,
        const float* __restrict__ h0,  float* __restrict__ HS,
        float* __restrict__ tail) {
    const int j = threadIdx.x;          // gate column 0..1023

    extern __shared__ unsigned char smem[];
    unsigned* wlds = (unsigned*)smem;                              // 311296 B
    float*    hsh  = (float*)(smem + WLDS_DWORDS * 4);             // 1024 B
    float*    zsh  = (float*)(smem + WLDS_DWORDS * 4 + F_DIM * 4); // 4096 B

    // --- init: pack Wh column j (coalesced along j) ---
    unsigned wreg[RREG_ROWS / 2];
    #pragma unroll
    for (int q = 0; q < RREG_ROWS / 2; ++q)
        wreg[q] = pack_bf16(Wh[(size_t)(2 * q) * G_DIM + j],
                            Wh[(size_t)(2 * q + 1) * G_DIM + j]);
    #pragma unroll
    for (int r = 0; r < RLDS_ROWS / 2; ++r)
        wlds[r * G_DIM + j] = pack_bf16(Wh[(size_t)(RREG_ROWS + 2 * r) * G_DIM + j],
                                        Wh[(size_t)(RREG_ROWS + 2 * r + 1) * G_DIM + j]);
    const float bj = bh[j];

    float c = 0.f;
    if (j < F_DIM) { c = c0[j]; hsh[j] = h0[j]; }
    __syncthreads();

    const float4* h4 = (const float4*)hsh;                 // rows 0..103 (26 chunks)
    const float2* h2 = (const float2*)(hsh + RREG_ROWS);   // rows 104..255 (76 pairs)
    float xnext = XWI[j];
    for (int t = 0; t < T_LEN; ++t) {
        float xcur = xnext;
        if (t + 1 < T_LEN) xnext = XWI[(size_t)(t + 1) * G_DIM + j];
        if (t + 2 < T_LEN) __builtin_prefetch(XWI + (size_t)(t + 2) * G_DIM + j, 0, 0);

        // 4 independent accumulator chains (64 MACs each)
        float a0 = bj + xcur, a1 = 0.f, a2 = 0.f, a3 = 0.f;

        // rows 0..103: weights in VGPRs (uniform-address broadcast h loads)
        #pragma unroll
        for (int m = 0; m < RREG_ROWS / 4; ++m) {   // 26 float4 chunks
            float4 hv = h4[m];
            mix_lo(a0, hv.x, wreg[2 * m]);
            mix_hi(a1, hv.y, wreg[2 * m]);
            mix_lo(a2, hv.z, wreg[2 * m + 1]);
            mix_hi(a3, hv.w, wreg[2 * m + 1]);
        }
        // rows 104..255: weights from LDS; 2 packed dwords / iteration so the
        // four accumulators rotate evenly
        #pragma unroll
        for (int r = 0; r < RLDS_ROWS / 4; ++r) {   // 38 iterations
            unsigned p0 = wlds[(2 * r) * G_DIM + j];
            unsigned p1 = wlds[(2 * r + 1) * G_DIM + j];
            float2 hp0 = h2[2 * r];
            float2 hp1 = h2[2 * r + 1];
            mix_lo(a0, hp0.x, p0);
            mix_hi(a1, hp0.y, p0);
            mix_lo(a2, hp1.x, p1);
            mix_hi(a3, hp1.y, p1);
        }

        zsh[j] = (a0 + a1) + (a2 + a3);
        __syncthreads();                 // all hsh reads done; zsh visible

        if (j < F_DIM) {
            float zi = zsh[j];
            float zf = zsh[j + F_DIM];
            float zg = zsh[j + 2 * F_DIM];
            float zo = zsh[j + 3 * F_DIM];
            float si = 1.f / (1.f + __expf(-zi));
            float sf = 1.f / (1.f + __expf(-zf));
            float tg = 2.f / (1.f + __expf(-2.f * zg)) - 1.f;
            float so = 1.f / (1.f + __expf(-zo));
            c = sf * c + si * tg;
            float h = so * (2.f / (1.f + __expf(-2.f * c)) - 1.f);
            hsh[j] = h;
            HS[(size_t)t * F_DIM + j] = h;
        }
        __syncthreads();                 // new h visible to all
    }

    if (j < F_DIM) {
        tail[j]             = c;         // cT
        tail[F_DIM + j]     = hsh[j];    // hT
        tail[2 * F_DIM + j] = c0[j];     // carry_c passthrough
        tail[3 * F_DIM + j] = h0[j];     // carry_h passthrough
    }
}

// ---------------------------------------------------------------------------
// Kernel 3: y = relu(layernorm(hs)); out = y @ Wd + bd.  One wave per row.
// ---------------------------------------------------------------------------
__global__ void ln_head(const float* __restrict__ HS, const float* __restrict__ scale,
                        const float* __restrict__ bias, const float* __restrict__ Wd,
                        const float* __restrict__ bd, float* __restrict__ OUT) {
    const int lane = threadIdx.x & 31;
    const int wid  = threadIdx.x >> 5;
    const int row  = blockIdx.x * 8 + wid;
    const float* x = HS + (size_t)row * F_DIM;

    float v[8];
    float s = 0.f, sq = 0.f;
    #pragma unroll
    for (int q = 0; q < 8; ++q) {
        v[q] = x[lane + 32 * q];
        s  += v[q];
        sq += v[q] * v[q];
    }
    #pragma unroll
    for (int off = 16; off > 0; off >>= 1) {
        s  += __shfl_xor(s,  off, 32);
        sq += __shfl_xor(sq, off, 32);
    }
    float mu  = s * (1.f / F_DIM);
    float var = sq * (1.f / F_DIM) - mu * mu;
    float rs  = __frsqrt_rn(var + 1e-6f);

    float o0 = 0.f, o1 = 0.f, o2 = 0.f;
    #pragma unroll
    for (int q = 0; q < 8; ++q) {
        int idx = lane + 32 * q;
        float y = (v[q] - mu) * rs * scale[idx] + bias[idx];
        y = fmaxf(y, 0.f);
        o0 += y * Wd[idx * 3 + 0];
        o1 += y * Wd[idx * 3 + 1];
        o2 += y * Wd[idx * 3 + 2];
    }
    #pragma unroll
    for (int off = 16; off > 0; off >>= 1) {
        o0 += __shfl_xor(o0, off, 32);
        o1 += __shfl_xor(o1, off, 32);
        o2 += __shfl_xor(o2, off, 32);
    }
    if (lane == 0) {
        OUT[(size_t)row * 3 + 0] = o0 + bd[0];
        OUT[(size_t)row * 3 + 1] = o1 + bd[1];
        OUT[(size_t)row * 3 + 2] = o2 + bd[2];
    }
}

// ---------------------------------------------------------------------------
extern "C" void kernel_launch(void* const* d_in, const int* in_sizes, int n_in,
                              void* d_out, int out_size, void* d_ws, size_t ws_size,
                              hipStream_t stream) {
    (void)in_sizes; (void)n_in; (void)out_size; (void)ws_size;
    const float* x   = (const float*)d_in[0];
    const float* c0  = (const float*)d_in[1];
    const float* h0  = (const float*)d_in[2];
    const float* Wi  = (const float*)d_in[3];
    const float* Wh  = (const float*)d_in[4];
    const float* bh  = (const float*)d_in[5];
    const float* lns = (const float*)d_in[6];
    const float* lnb = (const float*)d_in[7];
    const float* Wd  = (const float*)d_in[8];
    const float* bd  = (const float*)d_in[9];

    float* out  = (float*)d_out;
    float* xwi  = (float*)d_ws;                         // 32768*1024 f32 = 128 MB
    float* hs   = xwi + (size_t)T_LEN * G_DIM;          // 32768*256 f32 = 32 MB
    float* tail = out + (size_t)T_LEN * 3;              // cT | hT | carry_c | carry_h

    // 1) xWi = x @ Wi  (WMMA, bf16 split, f32 accumulate)
    {
        const int tiles  = (T_LEN / 16) * (G_DIM / 16);  // 131072 waves
        const int blocks = tiles / 8;                    // 8 waves / 256-thr block
        gemm_xwi_wmma<<<blocks, 256, 0, stream>>>(x, Wi, xwi);
    }
    // 2) sequential scan on a single WGP (Wh register+LDS resident, packed bf16)
    lstm_scan<<<1, 1024, SMEM_BYTES, stream>>>(xwi, Wh, bh, c0, h0, hs, tail);
    // 3) LayerNorm + ReLU + head
    ln_head<<<T_LEN / 8, 256, 0, stream>>>(hs, lns, lnb, Wd, bd, out);
}
